// GRDSE3HyperHyena_34308198761114
// MI455X (gfx1250) — matrix operand order repaired and checked by reading, not compiled, not run on hardware
//
#include <hip/hip_runtime.h>
#include <hip/hip_bf16.h>

// ---------------------------------------------------------------------------
// Problem constants
// ---------------------------------------------------------------------------
#define Bsz   64
#define Nseq  1024
#define Mmul  64
#define Pdim  64
#define Ddim  256          // 4*M
#define ROWS  (Bsz * Nseq) // 65536
#define EPSV  1e-5f

typedef __attribute__((ext_vector_type(16))) __bf16 v16bf;
typedef __attribute__((ext_vector_type(8)))  __bf16 bf16x8;
typedef __attribute__((ext_vector_type(8)))  float  v8f;

__device__ __forceinline__ __hip_bfloat16 f2bf(float f) { return __float2bfloat16(f); }
__device__ __forceinline__ float sigmoidf_(float x) { return 1.f / (1.f + expf(-x)); }

// ---------------------------------------------------------------------------
// 1. Sinusoidal PE + token embedding, then f @ Wf  -> fW [N, M]
// ---------------------------------------------------------------------------
__global__ void pe_feat_kernel(const float* __restrict__ tok_emb,
                               const float* __restrict__ Wf,
                               float* __restrict__ fW) {
  int n = blockIdx.x;          // 0..N-1
  int t = threadIdx.x;         // 0..63
  __shared__ float f[Pdim];
  {
    int p = t;
    int i2 = (p >> 1) * 2;
    float div = expf((float)i2 * (-logf(10000.f) / (float)Pdim));
    float ang = (float)n * div;
    float pe  = (p & 1) ? cosf(ang) : sinf(ang);
    int tt = (n == Nseq - 1) ? 2 : (n & 1);
    f[p] = pe + tok_emb[tt * Pdim + p];
  }
  __syncthreads();
  float acc = 0.f;
  #pragma unroll 8
  for (int p = 0; p < Pdim; ++p) acc += f[p] * Wf[p * Mmul + t];
  fW[n * Mmul + t] = acc;
}

// ---------------------------------------------------------------------------
// 2. Assemble h[b,n,4m+{0..3}] = {fW[n,m], x*wv}
// ---------------------------------------------------------------------------
__global__ void build_h_kernel(const float* __restrict__ x,
                               const float* __restrict__ wv,
                               const float* __restrict__ fW,
                               float* __restrict__ h) {
  size_t idx = (size_t)blockIdx.x * 256 + threadIdx.x; // rows*M
  if (idx >= (size_t)ROWS * Mmul) return;
  int m  = (int)(idx & 63);
  size_t rn = idx >> 6;              // b*N+n
  int n  = (int)(rn & (Nseq - 1));
  float* hr = h + rn * Ddim + 4 * m;
  float wm = wv[m];
  hr[0] = fW[n * Mmul + m];
  hr[1] = x[rn * 3 + 0] * wm;
  hr[2] = x[rn * 3 + 1] * wm;
  hr[3] = x[rn * 3 + 2] * wm;
}

// ---------------------------------------------------------------------------
// fp32 -> bf16 elementwise convert
// ---------------------------------------------------------------------------
__global__ void cvt_bf16_kernel(const float* __restrict__ src,
                                __hip_bfloat16* __restrict__ dst, size_t n) {
  size_t i = (size_t)blockIdx.x * 256 + threadIdx.x;
  if (i < n) dst[i] = f2bf(src[i]);
}

// Zero pad columns [H, Kp) of bf16 matrix with row stride Kp
__global__ void padzero_kernel(__hip_bfloat16* __restrict__ hb, int H, int Kp, int rows) {
  int pad = Kp - H;
  size_t total = (size_t)rows * pad;
  size_t idx = (size_t)blockIdx.x * 256 + threadIdx.x;
  if (idx >= total) return;
  size_t r = idx / pad;
  int c = (int)(idx % pad);
  hb[r * Kp + H + c] = f2bf(0.f);
}

// ---------------------------------------------------------------------------
// Weight convert: W [K x Ncols] f32 row-major -> WT [Np x Kp] bf16 (transposed,
// zero padded) so each WMMA B-fragment lane reads two runs of 8 contiguous bf16.
// Np is padded to a multiple of 64 (4 N-tiles per wave).
// ---------------------------------------------------------------------------
__global__ void wconv_kernel(const float* __restrict__ W,
                             __hip_bfloat16* __restrict__ WT,
                             int K, int Ncols, int Kp, int Np) {
  int idx = blockIdx.x * 256 + threadIdx.x;
  if (idx >= Np * Kp) return;
  int n = idx / Kp, k = idx % Kp;
  float v = (n < Ncols && k < K) ? W[(size_t)k * Ncols + n] : 0.f;
  WT[idx] = f2bf(v);
}

// ---------------------------------------------------------------------------
// WMMA bf16 GEMM with 2x4 register blocking:
//   C[rows x Ncols] (+)= A[rows x Kp] * WT^T
//   A  : bf16 row-major, K padded to Kp (mult of 32)
//   WT : bf16, [Np64 x Kp], row n = column n of W, Np64 mult of 64, zero padded
//   C  : fp32 row-major (leading dim Ncols)
// block = 256 threads = 8 waves; each wave owns a 32(M) x 64(N) C macro-tile
// (2 M-tiles x 4 N-tiles) -> 8 WMMAs per 12 fragment loads per k-step.
// grid = (Np64/64, rows/256)
// Fragment layouts per CDNA5 ISA 7.12.2 (wave32).
// ---------------------------------------------------------------------------
__global__ void __launch_bounds__(256)
wmma_gemm_kernel(const __hip_bfloat16* __restrict__ Abf,
                 const __hip_bfloat16* __restrict__ WTbf,
                 float* __restrict__ C,
                 int Kp, int Ncols, int accumulate) {
  const __bf16* A  = reinterpret_cast<const __bf16*>(Abf);
  const __bf16* WT = reinterpret_cast<const __bf16*>(WTbf);
  int lane = threadIdx.x & 31;
  int wave = threadIdx.x >> 5;
  int tm0 = (blockIdx.y * 8 + wave) * 2;  // first of 2 M tiles
  int tn0 = blockIdx.x * 4;               // first of 4 N tiles
  int hf   = lane >> 4;                   // lane half (0/1)
  int lrow = lane & 15;

  const __bf16* a0 = A + (size_t)(tm0 * 16 + lrow) * Kp;        // M tile 0 row
  const __bf16* a1 = a0 + (size_t)16 * Kp;                      // M tile 1 row
  const __bf16* bp[4];
  #pragma unroll
  for (int j = 0; j < 4; ++j)
    bp[j] = WT + (size_t)((tn0 + j) * 16 + lrow) * Kp;          // W column rows

  v8f c[2][4];
  #pragma unroll
  for (int i = 0; i < 2; ++i) {
    #pragma unroll
    for (int j = 0; j < 4; ++j) {
      int col = (tn0 + j) * 16 + lrow;
      if (accumulate && col < Ncols) {
        #pragma unroll
        for (int e = 0; e < 8; ++e)
          c[i][j][e] = C[(size_t)((tm0 + i) * 16 + e + 8 * hf) * Ncols + col];
      } else {
        #pragma unroll
        for (int e = 0; e < 8; ++e) c[i][j][e] = 0.f;
      }
    }
  }

  union U { v16bf v; bf16x8 q[2]; };
  for (int k0 = 0; k0 < Kp; k0 += 32) {
    // prefetch next k-chunk of the A rows (global_prefetch_b8)
    if (k0 + 32 < Kp) {
      __builtin_prefetch(a0 + k0 + 32, 0, 1);
      __builtin_prefetch(a1 + k0 + 32, 0, 1);
    }
    U ua[2];
    // A 16x32 bf16 fragment: lanes 0-15 -> K = k0+{0..7,16..23}; lanes 16-31 -> +8
    ua[0].q[0] = *reinterpret_cast<const bf16x8*>(a0 + k0 + 8 * hf);
    ua[0].q[1] = *reinterpret_cast<const bf16x8*>(a0 + k0 + 16 + 8 * hf);
    ua[1].q[0] = *reinterpret_cast<const bf16x8*>(a1 + k0 + 8 * hf);
    ua[1].q[1] = *reinterpret_cast<const bf16x8*>(a1 + k0 + 16 + 8 * hf);
    #pragma unroll
    for (int j = 0; j < 4; ++j) {
      // B 32x16 bf16 fragment: lanes 0-15 -> K = k0+{0..7,16..23}; lanes 16-31 -> +8
      U ub;
      ub.q[0] = *reinterpret_cast<const bf16x8*>(bp[j] + k0 + 8 * hf);
      ub.q[1] = *reinterpret_cast<const bf16x8*>(bp[j] + k0 + 16 + 8 * hf);
      c[0][j] = __builtin_amdgcn_wmma_f32_16x16x32_bf16(false, ua[0].v, false, ub.v,
                                                        (short)0, c[0][j], false, false);
      c[1][j] = __builtin_amdgcn_wmma_f32_16x16x32_bf16(false, ua[1].v, false, ub.v,
                                                        (short)0, c[1][j], false, false);
    }
  }

  #pragma unroll
  for (int i = 0; i < 2; ++i) {
    #pragma unroll
    for (int j = 0; j < 4; ++j) {
      int col = (tn0 + j) * 16 + lrow;
      if (col < Ncols) {
        #pragma unroll
        for (int e = 0; e < 8; ++e)
          C[(size_t)((tm0 + i) * 16 + e + 8 * hf) * Ncols + col] = c[i][j][e];
      }
    }
  }
}

// ---------------------------------------------------------------------------
// Elementwise multiply: a *= b
// ---------------------------------------------------------------------------
__global__ void mul_kernel(float* __restrict__ a, const float* __restrict__ b, size_t n) {
  size_t i = (size_t)blockIdx.x * 256 + threadIdx.x;
  if (i < n) a[i] *= b[i];
}

// ---------------------------------------------------------------------------
// In-LDS radix-2 1024-point complex FFT (256 threads).  sign=-1 fwd, +1 inv
// ---------------------------------------------------------------------------
__device__ __forceinline__ float2 cmulc(float2 a, float2 b) {
  return make_float2(a.x * b.x - a.y * b.y, a.x * b.y + a.y * b.x);
}

__device__ void fft1024_shared(float2* s, int tid, float sign) {
  // bit-reversal permutation (pairs disjoint; each swapped once)
  for (int i = tid; i < 1024; i += 256) {
    int j = (int)(__brev((unsigned)i) >> 22);
    if (j > i) { float2 t = s[i]; s[i] = s[j]; s[j] = t; }
  }
  __syncthreads();
  for (int len = 2; len <= 1024; len <<= 1) {
    int hl = len >> 1;
    float ang0 = sign * 6.283185307179586f / (float)len;
    for (int i = tid; i < 512; i += 256) {
      int blk = i / hl;
      int j   = i - blk * hl;
      int base = blk * len + j;
      float sn, cs;
      __sincosf(ang0 * (float)j, &sn, &cs);
      float2 w = make_float2(cs, sn);
      float2 u = s[base];
      float2 t = cmulc(w, s[base + hl]);
      s[base]      = make_float2(u.x + t.x, u.y + t.y);
      s[base + hl] = make_float2(u.x - t.x, u.y - t.y);
    }
    __syncthreads();
  }
}

// Forward DFT of filter column ch -> filtF[ch][0..N-1]
__global__ void filt_fft_kernel(const float* __restrict__ filt,
                                float2* __restrict__ filtF, int H) {
  __shared__ float2 sd[Nseq];
  int ch = blockIdx.x, tid = threadIdx.x;
  for (int n = tid; n < Nseq; n += 256)
    sd[n] = make_float2(filt[(size_t)n * H + ch], 0.f);
  __syncthreads();
  fft1024_shared(sd, tid, -1.f);
  for (int n = tid; n < Nseq; n += 256)
    filtF[(size_t)ch * Nseq + n] = sd[n];
}

// Circular conv of kv[b,:,ch] with filter (freq domain), gated by q, written
// as bf16 directly into the Wo-GEMM input (row stride KpH).
__global__ void conv_gate_kernel(const float* __restrict__ kv,
                                 const float* __restrict__ q,
                                 const float2* __restrict__ filtF,
                                 __hip_bfloat16* __restrict__ gout,
                                 int H, int KpH) {
  __shared__ float2 sd[Nseq];
  int b = blockIdx.x, ch = blockIdx.y, tid = threadIdx.x;
  for (int n = tid; n < Nseq; n += 256)
    sd[n] = make_float2(kv[((size_t)b * Nseq + n) * H + ch], 0.f);
  __syncthreads();
  fft1024_shared(sd, tid, -1.f);
  for (int n = tid; n < Nseq; n += 256)
    sd[n] = cmulc(sd[n], filtF[(size_t)ch * Nseq + n]);
  __syncthreads();
  fft1024_shared(sd, tid, 1.f);
  for (int n = tid; n < Nseq; n += 256) {
    float conv = sd[n].x * (1.f / (float)Nseq);
    size_t rn = (size_t)b * Nseq + n;
    float g = q[rn * H + ch] * conv;
    gout[rn * KpH + ch] = f2bf(g);
  }
}

// ---------------------------------------------------------------------------
// Split irreps of h into 4 bf16 matrices [rows x 64]: s, v0, v1, v2
// ---------------------------------------------------------------------------
__global__ void split_kernel(const float* __restrict__ h,
                             __hip_bfloat16* __restrict__ sb, int rows) {
  size_t idx = (size_t)blockIdx.x * 256 + threadIdx.x;
  if (idx >= (size_t)rows * Mmul) return;
  size_t r = idx >> 6;
  int m = (int)(idx & 63);
  const float* hr = h + r * Ddim + 4 * m;
  size_t st = (size_t)rows * Mmul;
  sb[idx]          = f2bf(hr[0]);
  sb[st + idx]     = f2bf(hr[1]);
  sb[2 * st + idx] = f2bf(hr[2]);
  sb[3 * st + idx] = f2bf(hr[3]);
}

// ---------------------------------------------------------------------------
// Batchnorm statistics: stats[m] = sum s^2, stats[64+m] = sum |v|^2 over rows
// ---------------------------------------------------------------------------
__global__ void zero_stats_kernel(float* stats) {
  if (threadIdx.x < 128) stats[threadIdx.x] = 0.f;
}

__global__ void stats_kernel(const float* __restrict__ ys,
                             const float* __restrict__ yv,
                             float* __restrict__ stats, int rows) {
  int tid = threadIdx.x;
  int c = tid & 63, sub = tid >> 6;          // 4 row-groups per block
  size_t st = (size_t)rows * Mmul;
  float ss = 0.f, vv = 0.f;
  for (int r = blockIdx.x * 4 + sub; r < rows; r += gridDim.x * 4) {
    size_t i = (size_t)r * Mmul + c;
    float s = ys[i];              ss += s * s;
    float a = yv[i];
    float b = yv[st + i];
    float d = yv[2 * st + i];
    vv += a * a + b * b + d * d;
  }
  __shared__ float sh[512];
  sh[tid] = ss; sh[256 + tid] = vv;
  __syncthreads();
  if (tid < 128) { sh[tid] += sh[tid + 128]; sh[256 + tid] += sh[256 + tid + 128]; }
  __syncthreads();
  if (tid < 64) {
    atomicAdd(&stats[tid],       sh[tid] + sh[tid + 64]);
    atomicAdd(&stats[64 + tid],  sh[256 + tid] + sh[256 + tid + 64]);
  }
}

// ---------------------------------------------------------------------------
// Fused batchnorm + norm-activation + residual into h
// ---------------------------------------------------------------------------
__global__ void normact_kernel(const float* __restrict__ ys,
                               const float* __restrict__ yv,
                               const float* __restrict__ stats,
                               const float* __restrict__ g0,
                               const float* __restrict__ g1,
                               float* __restrict__ h, int rows) {
  size_t idx = (size_t)blockIdx.x * 256 + threadIdx.x;
  if (idx >= (size_t)rows * Mmul) return;
  size_t r = idx >> 6;
  int m = (int)(idx & 63);
  float inv = 1.f / (float)rows;
  float n0 = sqrtf(stats[m] * inv + EPSV);
  float n1 = sqrtf(stats[64 + m] * inv + EPSV);
  size_t st = (size_t)rows * Mmul;
  float s  = ys[idx] / n0 * g0[m];
  float v0 = yv[idx] / n1 * g1[m];
  float v1 = yv[st + idx] / n1 * g1[m];
  float v2 = yv[2 * st + idx] / n1 * g1[m];
  float vn = sqrtf(v0 * v0 + v1 * v1 + v2 * v2 + EPSV);
  float gs = sigmoidf_(fabsf(s));
  float gv = sigmoidf_(vn);
  float* hr = h + r * Ddim + 4 * m;
  hr[0] += s * gs;
  hr[1] += v0 * gv;
  hr[2] += v1 * gv;
  hr[3] += v2 * gv;
}

// ---------------------------------------------------------------------------
// Mean-pool over tokens + contract vector irreps with w_out -> out [B,3]
// ---------------------------------------------------------------------------
__global__ void pool_kernel(const float* __restrict__ h,
                            const float* __restrict__ w_out,
                            float* __restrict__ out) {
  int b = blockIdx.x, tid = threadIdx.x;
  float a0 = 0.f, a1 = 0.f, a2 = 0.f;
  for (int n = tid; n < Nseq; n += 256) {
    const float* row = h + ((size_t)b * Nseq + n) * Ddim;
    #pragma unroll 4
    for (int m = 0; m < Mmul; ++m) {
      float w = w_out[m];
      a0 += row[4 * m + 1] * w;
      a1 += row[4 * m + 2] * w;
      a2 += row[4 * m + 3] * w;
    }
  }
  __shared__ float sh[3][256];
  sh[0][tid] = a0; sh[1][tid] = a1; sh[2][tid] = a2;
  __syncthreads();
  for (int off = 128; off > 0; off >>= 1) {
    if (tid < off) {
      sh[0][tid] += sh[0][tid + off];
      sh[1][tid] += sh[1][tid + off];
      sh[2][tid] += sh[2][tid + off];
    }
    __syncthreads();
  }
  if (tid == 0) {
    out[b * 3 + 0] = sh[0][0] / (float)Nseq;
    out[b * 3 + 1] = sh[1][0] / (float)Nseq;
    out[b * 3 + 2] = sh[2][0] / (float)Nseq;
  }
}

// ---------------------------------------------------------------------------
// Host orchestration
// ---------------------------------------------------------------------------
static inline unsigned cdivu(size_t a, size_t b) { return (unsigned)((a + b - 1) / b); }

extern "C" void kernel_launch(void* const* d_in, const int* in_sizes, int n_in,
                              void* d_out, int out_size, void* d_ws, size_t ws_size,
                              hipStream_t stream) {
  (void)in_sizes; (void)n_in; (void)out_size; (void)ws_size;
  const float* x       = (const float*)d_in[0];
  const float* tok_emb = (const float*)d_in[1];
  const float* Wf      = (const float*)d_in[2];
  const float* wv      = (const float*)d_in[3];
  const float* w_out   = (const float*)d_in[4];

  char* ws = (char*)d_ws;
  const size_t MiB = 1ull << 20;
  float*          h     = (float*)(ws);                       //  64 MiB  h [rows,256]
  __hip_bfloat16* hb    = (__hip_bfloat16*)(ws + 64  * MiB);  //  48 MiB  bf16 GEMM-A buffer
  float*          qb    = (float*)(ws + 112 * MiB);           //  96 MiB  q / (ys,yv)
  float*          kb    = (float*)(ws + 208 * MiB);           //  96 MiB  k -> kv
  float*          vb    = (float*)(ws + 304 * MiB);           //  96 MiB  v
  __hip_bfloat16* wq_t  = (__hip_bfloat16*)(ws + 400 * MiB);
  __hip_bfloat16* wk_t  = (__hip_bfloat16*)(ws + 400 * MiB + 256 * 1024);
  __hip_bfloat16* wv_t  = (__hip_bfloat16*)(ws + 400 * MiB + 512 * 1024);
  __hip_bfloat16* wo_t  = (__hip_bfloat16*)(ws + 400 * MiB + 768 * 1024);
  __hip_bfloat16* wm0_t = (__hip_bfloat16*)(ws + 401 * MiB);
  __hip_bfloat16* wm1_t = (__hip_bfloat16*)(ws + 401 * MiB + 64 * 1024);
  float2*         filtF = (float2*)(ws + 402 * MiB);          //   3 MiB
  float*          stats = (float*)(ws + 405 * MiB);
  float*          fW    = (float*)(ws + 405 * MiB + 8192);    // 256 KiB

  const int rows = ROWS;

  // input featurization
  pe_feat_kernel<<<Nseq, Pdim, 0, stream>>>(tok_emb, Wf, fW);
  build_h_kernel<<<cdivu((size_t)rows * Mmul, 256), 256, 0, stream>>>(x, wv, fW, h);

  const int Hs[3] = {360, 360, 160};
  for (int L = 0; L < 3; ++L) {
    const float* Wq  = (const float*)d_in[5 + 9 * L + 0];
    const float* Wk  = (const float*)d_in[5 + 9 * L + 1];
    const float* Wv_ = (const float*)d_in[5 + 9 * L + 2];
    const float* ft  = (const float*)d_in[5 + 9 * L + 3];
    const float* Wo  = (const float*)d_in[5 + 9 * L + 4];
    const float* Wm0 = (const float*)d_in[5 + 9 * L + 5];
    const float* Wm1 = (const float*)d_in[5 + 9 * L + 6];
    const float* g0  = (const float*)d_in[5 + 9 * L + 7];
    const float* g1  = (const float*)d_in[5 + 9 * L + 8];
    const int H    = Hs[L];
    const int Np64 = ((H + 63) / 64) * 64;  // 384 / 192  (N padded to 4-tile blocks)
    const int KpH  = ((H + 31) / 32) * 32;  // 384 / 160  (K padded to WMMA depth)

    // weight conversion (transpose + pad to bf16)
    wconv_kernel<<<cdivu((size_t)Np64 * Ddim, 256), 256, 0, stream>>>(Wq,  wq_t,  Ddim, H, Ddim, Np64);
    wconv_kernel<<<cdivu((size_t)Np64 * Ddim, 256), 256, 0, stream>>>(Wk,  wk_t,  Ddim, H, Ddim, Np64);
    wconv_kernel<<<cdivu((size_t)Np64 * Ddim, 256), 256, 0, stream>>>(Wv_, wv_t,  Ddim, H, Ddim, Np64);
    wconv_kernel<<<cdivu((size_t)Ddim * KpH, 256), 256, 0, stream>>>(Wo, wo_t, H, Ddim, KpH, Ddim);
    wconv_kernel<<<cdivu((size_t)Mmul * Mmul, 256), 256, 0, stream>>>(Wm0, wm0_t, Mmul, Mmul, Mmul, Mmul);
    wconv_kernel<<<cdivu((size_t)Mmul * Mmul, 256), 256, 0, stream>>>(Wm1, wm1_t, Mmul, Mmul, Mmul, Mmul);

    // h -> bf16 (Kp = 256)
    cvt_bf16_kernel<<<cdivu((size_t)rows * Ddim, 256), 256, 0, stream>>>(h, hb, (size_t)rows * Ddim);

    // q,k,v projections (WMMA, 2x4-blocked)
    dim3 gqkv(Np64 / 64, rows / 256);
    wmma_gemm_kernel<<<gqkv, 256, 0, stream>>>(hb, wq_t, qb, Ddim, H, 0);
    wmma_gemm_kernel<<<gqkv, 256, 0, stream>>>(hb, wk_t, kb, Ddim, H, 0);
    wmma_gemm_kernel<<<gqkv, 256, 0, stream>>>(hb, wv_t, vb, Ddim, H, 0);

    // kv = k * v (in place in kb)
    mul_kernel<<<cdivu((size_t)rows * H, 256), 256, 0, stream>>>(kb, vb, (size_t)rows * H);

    // filter spectrum
    filt_fft_kernel<<<H, 256, 0, stream>>>(ft, filtF, H);

    // gated circular conv -> bf16 GEMM input (row stride KpH)
    if (KpH > H)
      padzero_kernel<<<cdivu((size_t)rows * (KpH - H), 256), 256, 0, stream>>>(hb, H, KpH, rows);
    conv_gate_kernel<<<dim3(Bsz, H), 256, 0, stream>>>(kb, qb, filtF, hb, H, KpH);

    // h += gated @ Wo  (WMMA, accumulator preloaded from h)
    dim3 gwo(Ddim / 64, rows / 256);
    wmma_gemm_kernel<<<gwo, 256, 0, stream>>>(hb, wo_t, h, KpH, Ddim, 1);

    // reg_linear: split irreps, 4 small WMMA GEMMs
    split_kernel<<<cdivu((size_t)rows * Mmul, 256), 256, 0, stream>>>(h, hb, rows);
    size_t st = (size_t)rows * Mmul;
    dim3 gml(Mmul / 64, rows / 256);
    wmma_gemm_kernel<<<gml, 256, 0, stream>>>(hb,          wm0_t, qb,          Mmul, Mmul, 0);
    wmma_gemm_kernel<<<gml, 256, 0, stream>>>(hb + st,     wm1_t, qb + st,     Mmul, Mmul, 0);
    wmma_gemm_kernel<<<gml, 256, 0, stream>>>(hb + 2 * st, wm1_t, qb + 2 * st, Mmul, Mmul, 0);
    wmma_gemm_kernel<<<gml, 256, 0, stream>>>(hb + 3 * st, wm1_t, qb + 3 * st, Mmul, Mmul, 0);

    // batchnorm stats + fused norm/activation/residual
    zero_stats_kernel<<<1, 128, 0, stream>>>(stats);
    stats_kernel<<<256, 256, 0, stream>>>(qb, qb + st, stats, rows);
    normact_kernel<<<cdivu((size_t)rows * Mmul, 256), 256, 0, stream>>>(
        qb, qb + st, stats, g0, g1, h, rows);
  }

  // pooling + output contraction
  pool_kernel<<<Bsz, 256, 0, stream>>>(h, w_out, (float*)d_out);
}
